// GCN_5935644803541
// MI455X (gfx1250) — matrix operand order repaired and checked by reading, not compiled, not run on hardware
//
#include <hip/hip_runtime.h>
#include <math.h>

#define NNODES 50000
#define NEDGES 800000
#define NE2    (NEDGES + NNODES)   // edges + self loops
#define NGRAPH 64
#define NCLS   8
#define HIDW   64
#define SLOPE  0.2f

typedef __attribute__((ext_vector_type(16))) _Float16 v16h;
typedef __attribute__((ext_vector_type(8)))  float    v8f;

// ---------------------------------------------------------------- utilities
__global__ void fill_kernel(float* __restrict__ p, float v, int n) {
    int i = blockIdx.x * blockDim.x + threadIdx.x;
    if (i < n) p[i] = v;
}

__device__ inline void atomicMaxF(float* addr, float v) {
    // order-preserving bit tricks: signed-max for v>=0, unsigned-min for v<0
    if (v >= 0.0f) atomicMax((int*)addr, __float_as_int(v));
    else           atomicMin((unsigned int*)addr, __float_as_uint(v));
}

__device__ inline void edge_nodes(const int* __restrict__ ei, int e, int& s, int& d) {
    if (e < NEDGES) { s = ei[e]; d = ei[NEDGES + e]; }
    else            { s = e - NEDGES; d = s; }        // self loop
}

// ---------------------------------------------------------------- WMMA GEMM
// H[N x OUT] = X[N x K] * W[K x OUT]; K <= 64, OUT padded to NT*16.
// One wave = 16-row strip x NT 16-col tiles; 8 waves/block -> 128 rows/block.
// W is staged to LDS pre-swizzled into WMMA B-fragment order so each lane
// fetches its 16 halfs as one aligned v16h (2x ds_load_b128).
// FULLK: K is a multiple of 32 -> A loads are unguarded float4s.
template <int NT, bool FULLK>
__global__ __launch_bounds__(256) void gemm_wmma_kernel(
    const float* __restrict__ X, const float* __restrict__ W,
    float* __restrict__ H, int Nrows, int K, int OUT) {
    // [2 ksteps][NT tiles][32 lanes] fragments of 16 halfs
    __shared__ v16h Wl[2 * NT * 32];
    _Float16* Wlh = (_Float16*)Wl;
    for (int i = threadIdx.x; i < 2 * NT * 512; i += 256) {
        int j    = i & 15;           // half index within fragment
        int l    = (i >> 4) & 31;    // lane
        int tile = i >> 9;           // kstep*NT + nt
        int kstep = tile / NT, nt = tile % NT;
        int k = kstep * 32 + (l >> 4) * 8 + (j < 8 ? j : j + 8);   // B: lanes 0-15 K=0-7,16-23; 16-31 K=8-15,24-31
        int n = nt * 16 + (l & 15);
        float v = (k < K && n < OUT) ? W[k * OUT + n] : 0.0f;
        Wlh[i] = (_Float16)v;
    }
    __syncthreads();

    int wid  = threadIdx.x >> 5;
    int lane = threadIdx.x & 31;
    int rowBase = blockIdx.x * 128 + wid * 16;
    if (rowBase >= Nrows) return;
    int ml = lane & 15;       // row (A) / col (B,C,D) within tile
    int kh = lane >> 4;       // lane-half selector

    int rv = rowBase + ml;
    if (rv > Nrows - 1) rv = Nrows - 1;            // clamp: keeps loads vectorizable
    const float* Xr = X + (size_t)rv * K;

    v8f acc[NT] = {};
#pragma unroll
    for (int kstep = 0; kstep < 2; ++kstep) {
        if (kstep * 32 < K) {
            // A fragment: 16x32 f16; lane's K-slice = [kb..kb+7] and [kb+16..kb+23]
            v16h a;
            int kb = kstep * 32 + kh * 8;
            if (FULLK) {
                float4 p0 = *(const float4*)(Xr + kb);
                float4 p1 = *(const float4*)(Xr + kb + 4);
                float4 p2 = *(const float4*)(Xr + kb + 16);
                float4 p3 = *(const float4*)(Xr + kb + 20);
                a[0]  = (_Float16)p0.x;  a[1]  = (_Float16)p0.y;
                a[2]  = (_Float16)p0.z;  a[3]  = (_Float16)p0.w;
                a[4]  = (_Float16)p1.x;  a[5]  = (_Float16)p1.y;
                a[6]  = (_Float16)p1.z;  a[7]  = (_Float16)p1.w;
                a[8]  = (_Float16)p2.x;  a[9]  = (_Float16)p2.y;
                a[10] = (_Float16)p2.z;  a[11] = (_Float16)p2.w;
                a[12] = (_Float16)p3.x;  a[13] = (_Float16)p3.y;
                a[14] = (_Float16)p3.z;  a[15] = (_Float16)p3.w;
            } else {
#pragma unroll
                for (int j = 0; j < 16; ++j) {
                    int k = kb + (j < 8 ? j : j + 8);
                    a[j] = (_Float16)((k < K) ? Xr[k] : 0.0f);
                }
            }
#pragma unroll
            for (int nt = 0; nt < NT; ++nt) {
                v16h b = Wl[(kstep * NT + nt) * 32 + lane];
                acc[nt] = __builtin_amdgcn_wmma_f32_16x16x32_f16(
                    false, a, false, b, (short)0, acc[nt], false, false);
            }
        }
    }
#pragma unroll
    for (int nt = 0; nt < NT; ++nt) {
#pragma unroll
        for (int r = 0; r < 8; ++r) {
            int mrow = rowBase + r + 8 * kh;   // D layout: M = r + 8*kh
            int col  = nt * 16 + ml;
            if (mrow < Nrows && col < OUT) H[mrow * OUT + col] = acc[nt][r];
        }
    }
}

// ---------------------------------------------------------------- attention
__global__ void alphas_kernel(const float* __restrict__ H,
                              const float* __restrict__ aw,
                              const float* __restrict__ dw,
                              float* __restrict__ asv, float* __restrict__ adv,
                              int n, int OUT) {
    int i = blockIdx.x * blockDim.x + threadIdx.x;
    if (i >= n) return;
    const float4* Hr = (const float4*)(H + (size_t)i * OUT);   // rows 32B aligned
    const float4* A4 = (const float4*)aw;
    const float4* D4 = (const float4*)dw;
    float s = 0.0f, d = 0.0f;
    for (int f = 0; f < OUT / 4; ++f) {
        float4 h = Hr[f], a = A4[f], b = D4[f];
        s += h.x * a.x + h.y * a.y + h.z * a.z + h.w * a.w;
        d += h.x * b.x + h.y * b.y + h.z * b.z + h.w * b.w;
    }
    asv[i] = s;
    adv[i] = d;
}

__global__ void edge_max_kernel(const int* __restrict__ ei,
                                const float* __restrict__ asv,
                                const float* __restrict__ adv,
                                float* __restrict__ mM) {
    int e = blockIdx.x * blockDim.x + threadIdx.x;
    if (e >= NE2) return;
    int s, d; edge_nodes(ei, e, s, d);
    float ev = asv[s] + adv[d];
    ev = ev > 0.0f ? ev : SLOPE * ev;       // leaky relu
    atomicMaxF(&mM[d], ev);
}

__global__ void edge_exp_kernel(const int* __restrict__ ei,
                                const float* __restrict__ asv,
                                const float* __restrict__ adv,
                                const float* __restrict__ mM,
                                float* __restrict__ exv,
                                float* __restrict__ den) {
    int e = blockIdx.x * blockDim.x + threadIdx.x;
    if (e >= NE2) return;
    int s, d; edge_nodes(ei, e, s, d);
    float ev = asv[s] + adv[d];
    ev = ev > 0.0f ? ev : SLOPE * ev;
    float xv = __expf(ev - mM[d]);
    exv[e] = xv;
    atomicAdd(&den[d], xv);
}

__global__ void edge_agg_kernel(const int* __restrict__ ei,
                                const float* __restrict__ exv,
                                const float* __restrict__ den,
                                const float* __restrict__ H,
                                float* __restrict__ AGG,
                                int lg, int total) {
    int idx = blockIdx.x * blockDim.x + threadIdx.x;
    if (idx >= total) return;
    int e = idx >> lg;
    int f = idx & ((1 << lg) - 1);
    int s, d; edge_nodes(ei, e, s, d);
    float alpha = exv[e] / den[d];
    atomicAdd(&AGG[(d << lg) + f], alpha * H[(s << lg) + f]);
}

__global__ void bias_relu_kernel(float* __restrict__ AGG,
                                 const float* __restrict__ bias,
                                 int total, int fmask) {
    int i = blockIdx.x * blockDim.x + threadIdx.x;
    if (i >= total) return;
    float v = AGG[i] + bias[i & fmask];
    AGG[i] = v > 0.0f ? v : 0.0f;
}

// ---------------------------------------------------------------- pooling
__global__ void count_kernel(const int* __restrict__ batch, float* __restrict__ cnt) {
    int i = blockIdx.x * blockDim.x + threadIdx.x;
    if (i < NNODES) atomicAdd(&cnt[batch[i]], 1.0f);
}

__global__ void pool_kernel(const int* __restrict__ batch,
                            const float* __restrict__ h3,
                            float* __restrict__ pooled) {
    int idx = blockIdx.x * blockDim.x + threadIdx.x;
    if (idx >= NNODES * NCLS) return;
    int i = idx >> 3, f = idx & 7;
    atomicAdd(&pooled[batch[i] * NCLS + f], h3[i * NCLS + f]);
}

__global__ void final_kernel(const float* __restrict__ pooled,
                             const float* __restrict__ cnt,
                             float* __restrict__ out) {
    int idx = blockIdx.x * blockDim.x + threadIdx.x;
    if (idx >= NGRAPH * NCLS) return;
    float c = fmaxf(cnt[idx >> 3], 1.0f);
    float v = pooled[idx] / c;
    out[idx] = 1.0f / (1.0f + __expf(-v));
}

// ---------------------------------------------------------------- launcher
extern "C" void kernel_launch(void* const* d_in, const int* in_sizes, int n_in,
                              void* d_out, int out_size, void* d_ws, size_t ws_size,
                              hipStream_t stream) {
    (void)in_sizes; (void)n_in; (void)out_size; (void)ws_size;
    const float* x     = (const float*)d_in[0];
    const int*   ei    = (const int*)d_in[1];
    // d_in[2] edge_attr: unused by reference
    const int*   batch = (const int*)d_in[3];
    const float* W1 = (const float*)d_in[4],  *a1s = (const float*)d_in[5];
    const float* a1d = (const float*)d_in[6], *b1  = (const float*)d_in[7];
    const float* W2 = (const float*)d_in[8],  *a2s = (const float*)d_in[9];
    const float* a2d = (const float*)d_in[10],*b2  = (const float*)d_in[11];
    const float* W3 = (const float*)d_in[12], *a3s = (const float*)d_in[13];
    const float* a3d = (const float*)d_in[14],*b3  = (const float*)d_in[15];
    float* out = (float*)d_out;

    float* ws     = (float*)d_ws;
    float* hA     = ws;                                  // N*64
    float* hB     = hA  + (size_t)NNODES * HIDW;         // N*64
    float* h3     = hB  + (size_t)NNODES * HIDW;         // N*8
    float* asv    = h3  + (size_t)NNODES * NCLS;         // N
    float* adv    = asv + NNODES;                        // N
    float* mM     = adv + NNODES;                        // N
    float* den    = mM  + NNODES;                        // N
    float* exv    = den + NNODES;                        // NE2
    float* pooled = exv + NE2;                           // 64*8
    float* cnt    = pooled + NGRAPH * NCLS;              // 64

    dim3 blk(256);
    auto g = [](int n) { return dim3((unsigned)((n + 255) / 256)); };

    auto layer = [&](const float* Xin, int K, int OUT, int NT, bool fullk, int lg,
                     const float* W, const float* aw, const float* dw,
                     const float* bias, float* H, float* AGG) {
        dim3 gemmGrid((unsigned)((NNODES + 127) / 128));
        if (NT == 4) {
            if (fullk) gemm_wmma_kernel<4, true ><<<gemmGrid, blk, 0, stream>>>(Xin, W, H, NNODES, K, OUT);
            else       gemm_wmma_kernel<4, false><<<gemmGrid, blk, 0, stream>>>(Xin, W, H, NNODES, K, OUT);
        } else {
            gemm_wmma_kernel<1, true><<<gemmGrid, blk, 0, stream>>>(Xin, W, H, NNODES, K, OUT);
        }
        alphas_kernel<<<g(NNODES), blk, 0, stream>>>(H, aw, dw, asv, adv, NNODES, OUT);
        fill_kernel<<<g(NNODES), blk, 0, stream>>>(mM, -INFINITY, NNODES);
        fill_kernel<<<g(NNODES), blk, 0, stream>>>(den, 0.0f, NNODES);
        fill_kernel<<<g(NNODES * OUT), blk, 0, stream>>>(AGG, 0.0f, NNODES * OUT);
        edge_max_kernel<<<g(NE2), blk, 0, stream>>>(ei, asv, adv, mM);
        edge_exp_kernel<<<g(NE2), blk, 0, stream>>>(ei, asv, adv, mM, exv, den);
        int total = NE2 << lg;
        edge_agg_kernel<<<g(total), blk, 0, stream>>>(ei, exv, den, H, AGG, lg, total);
        bias_relu_kernel<<<g(NNODES * OUT), blk, 0, stream>>>(AGG, bias, NNODES * OUT, OUT - 1);
    };

    // layer1: x[N,11] -> hB[N,64]   (gemm result in hA)
    layer(x, 11, HIDW, 4, false, 6, W1, a1s, a1d, b1, hA, hB);
    // layer2: hB -> hB (gemm result in hA; hB free after gemm+alphas read it)
    layer(hB, HIDW, HIDW, 4, true, 6, W2, a2s, a2d, b2, hA, hB);
    // layer3: hB -> h3[N,8] (gemm result in hA, 8-wide rows)
    layer(hB, HIDW, NCLS, 1, true, 3, W3, a3s, a3d, b3, hA, h3);

    fill_kernel<<<g(NGRAPH * NCLS), blk, 0, stream>>>(pooled, 0.0f, NGRAPH * NCLS);
    fill_kernel<<<g(NGRAPH), blk, 0, stream>>>(cnt, 0.0f, NGRAPH);
    count_kernel<<<g(NNODES), blk, 0, stream>>>(batch, cnt);
    pool_kernel<<<g(NNODES * NCLS), blk, 0, stream>>>(batch, h3, pooled);
    final_kernel<<<g(NGRAPH * NCLS), blk, 0, stream>>>(pooled, cnt, out);
}
